// MambaJSCC_Encoder_30520037605579
// MI455X (gfx1250) — matrix operand order repaired and checked by reading, not compiled, not run on hardware
//
#include <hip/hip_runtime.h>

#define B_   64
#define L_   1024
#define E_   320
#define C_   32
#define H_   480
#define HP_  512            // H padded to 4*16*8 tiles (branch-free inner loop)
#define CP_  64             // C padded to 4*16
#define LN_  7
#define TM   32
#define NTOK (B_ * L_)
#define EPSF 1e-5f

typedef __attribute__((ext_vector_type(16))) __bf16 v16bf;
typedef __attribute__((ext_vector_type(8)))  __bf16 v8bf;
typedef __attribute__((ext_vector_type(8)))  float  v8f;

__device__ __forceinline__ unsigned short f2bf_bits(float f) {
  unsigned int u = __float_as_uint(f);
  u += 0x7FFFu + ((u >> 16) & 1u);   // round-to-nearest-even
  return (unsigned short)(u >> 16);
}

// ---------------------------------------------------------------------------
// Branch-free WMMA tile loop.  A: bf16 LDS tile (row-major, lda elems).
// W: bf16 global (L2-resident), row-major, out-dim padded so that all
// MAXJ tiles of every col-group are valid -> no guards around WMMA.
// Per k-step: issue ALL B-fragment loads, then ALL WMMAs (overlappable).
// ---------------------------------------------------------------------------
template<int MAXJ>
__device__ __forceinline__ void gemm_run(const unsigned short* Abf, int lda,
                                         const unsigned short* W, int ldw,
                                         int ksteps, v8f* acc,
                                         int mblk, int colgrp, int lane)
{
  const int rsel  = lane & 15;
  const int khalf = lane >> 4;
  const int arow  = mblk * 16 + rsel;
  const unsigned short* ap = Abf + (size_t)arow * lda + khalf * 8;
  const unsigned short* wp[MAXJ];
  #pragma unroll
  for (int j = 0; j < MAXJ; ++j)
    wp[j] = W + (size_t)((colgrp + 4 * j) * 16 + rsel) * ldw + khalf * 8;

  for (int kt = 0; kt < ksteps; ++kt) {
    union { v16bf v; v8bf h[2]; } ua;
    ua.h[0] = *(const v8bf*)(ap);        // LDS: K 0-7 / 8-15 half
    ua.h[1] = *(const v8bf*)(ap + 16);   // LDS: K 16-23 / 24-31 half

    v16bf bf[MAXJ];
    #pragma unroll
    for (int j = 0; j < MAXJ; ++j) {     // load group: MAXJ*2 b128 loads
      union { v16bf v; v8bf h[2]; } ub;
      ub.h[0] = *(const v8bf*)(wp[j]);
      ub.h[1] = *(const v8bf*)(wp[j] + 16);
      bf[j] = ub.v;
    }
    #pragma unroll
    for (int j = 0; j < MAXJ; ++j)       // wmma group
      acc[j] = __builtin_amdgcn_wmma_f32_16x16x32_bf16(
          false, ua.v, false, bf[j], (short)0, acc[j], false, false);

    ap += 32;
    #pragma unroll
    for (int j = 0; j < MAXJ; ++j) wp[j] += 32;
  }
}

// ---------------------------------------------------------------------------
// fp32 -> bf16 weight conversion with out-dim zero padding.
// dst is nmat matrices of (rows_pad x cols); src is nmat x (rows x cols).
// ---------------------------------------------------------------------------
__global__ void cvt_pad_kernel(const float* __restrict__ src,
                               unsigned short* __restrict__ dst,
                               int rows, int rows_pad, int cols, int nmat) {
  int i = blockIdx.x * blockDim.x + threadIdx.x;
  int per = rows_pad * cols;
  if (i >= nmat * per) return;
  int m   = i / per;
  int rem = i - m * per;
  int r   = rem / cols;
  int c   = rem - r * cols;
  unsigned short v = 0;
  if (r < rows) v = f2bf_bits(src[((size_t)m * rows + r) * cols + c]);
  dst[i] = v;
}

// ---------------------------------------------------------------------------
// SNR modulator: snr is one scalar broadcast over the batch, so the 7
// sigmoid gate vectors (7 x 480) are batch-invariant.  One block.
// ---------------------------------------------------------------------------
__global__ void modulator_kernel(const float* __restrict__ snr,
                                 const float* __restrict__ w1, const float* __restrict__ b1,
                                 const float* __restrict__ w2, const float* __restrict__ b2,
                                 const float* __restrict__ w3, const float* __restrict__ b3,
                                 float* __restrict__ bm_out) {
  __shared__ float h1[H_];
  __shared__ float h2[H_];
  const int t = threadIdx.x;
  const float s0 = snr[0];
  for (int i = 0; i < LN_; ++i) {
    if (t < H_) h1[t] = fmaxf(0.f, s0 * w1[i * H_ + t] + b1[i * H_ + t]);
    __syncthreads();
    if (t < H_) {
      float s = b2[i * H_ + t];
      const float* wr = w2 + ((size_t)i * H_ + t) * H_;
      for (int k = 0; k < H_; ++k) s += wr[k] * h1[k];
      h2[t] = fmaxf(0.f, s);
    }
    __syncthreads();
    if (t < H_) {
      float s = b3[i * H_ + t];
      const float* wr = w3 + ((size_t)i * H_ + t) * H_;
      for (int k = 0; k < H_; ++k) s += wr[k] * h2[k];
      bm_out[i * H_ + t] = 1.f / (1.f + __expf(-s));
    }
    __syncthreads();
  }
}

// ---------------------------------------------------------------------------
// Fused: LayerNorm -> 7-layer gated MLP chain -> sigmoid mod -> head GEMM.
// 32 tokens per workgroup; all activations live in LDS (~100 KB of 320 KB).
// ---------------------------------------------------------------------------
__global__ __launch_bounds__(256)
void fused_encoder_kernel(const float* __restrict__ x,
                          const float* __restrict__ norm_w,
                          const float* __restrict__ norm_b,
                          const unsigned short* __restrict__ w0b,
                          const float* __restrict__ sm0_b,
                          const unsigned short* __restrict__ wmidb,
                          const float* __restrict__ sm_mid_b,
                          const unsigned short* __restrict__ wlastb,
                          const float* __restrict__ sm_last_b,
                          const unsigned short* __restrict__ wheadb,
                          const float* __restrict__ head_b,
                          const float* __restrict__ bm,
                          float* __restrict__ out)
{
  __shared__ __align__(16) float          s_xn[TM][E_];   // 40 KB f32 normed x
  __shared__ __align__(16) unsigned short s_tA[TM][H_];   // 30 KB bf16 ping
  __shared__ __align__(16) unsigned short s_tB[TM][H_];   // 30 KB bf16 pong

  const int tid    = threadIdx.x;
  const int lane   = tid & 31;
  const int wave   = __builtin_amdgcn_readfirstlane(tid >> 5);  // scalar
  const int mblk   = wave & 1;    // row block (2 x 16 rows)
  const int colgrp = wave >> 1;   // column phase (0..3)
  const int t0     = blockIdx.x * TM;

  // ---------------- LayerNorm; write xn f32 + xn bf16 (into s_tB) ----------
  #pragma unroll
  for (int rr = 0; rr < TM / 8; ++rr) {
    const int row = wave * (TM / 8) + rr;
    const float* xr = x + (size_t)(t0 + row) * E_;
    float v[E_ / 32];
    float s = 0.f, ss = 0.f;
    #pragma unroll
    for (int q = 0; q < E_ / 32; ++q) {
      float f = xr[lane + 32 * q];
      v[q] = f; s += f; ss += f * f;
    }
    #pragma unroll
    for (int off = 16; off >= 1; off >>= 1) {
      s  += __shfl_xor(s, off, 32);
      ss += __shfl_xor(ss, off, 32);
    }
    const float mu   = s * (1.f / E_);
    const float var  = ss * (1.f / E_) - mu * mu;
    const float rstd = rsqrtf(var + EPSF);
    #pragma unroll
    for (int q = 0; q < E_ / 32; ++q) {
      const int e = lane + 32 * q;
      const float xnv = (v[q] - mu) * rstd * norm_w[e] + norm_b[e];
      s_xn[row][e] = xnv;
      s_tB[row][e] = f2bf_bits(xnv);   // bf16 A operand for GEMM0
    }
  }
  __syncthreads();

  const int rsel  = lane & 15;
  const int khalf = lane >> 4;
  const int rbase = mblk * 16 + khalf * 8;

  v8f acc[8];

  // ---------------- GEMM0: xn @ sm0_w.T, + bias, x bm[0] -------------------
  #pragma unroll
  for (int j = 0; j < 8; ++j)
    #pragma unroll
    for (int r = 0; r < 8; ++r) acc[j][r] = 0.f;
  gemm_run<8>(&s_tB[0][0], H_, w0b, E_, E_/32, acc, mblk, colgrp, lane);
  #pragma unroll
  for (int j = 0; j < 8; ++j) {
    const int nb = colgrp + 4 * j;
    if (nb < H_/16) {                        // skip zero-padded columns
      const int colb = nb * 16 + rsel;
      const float bias = sm0_b[colb];
      const float g    = bm[colb];
      #pragma unroll
      for (int r = 0; r < 8; ++r)
        s_tA[rbase + r][colb] = f2bf_bits((acc[j][r] + bias) * g);
    }
  }
  __syncthreads();

  // ---------------- Mid layers 1..6: temp @ W.T, + bias, x bm[i] -----------
  const unsigned short* src = &s_tA[0][0];
  unsigned short*       dst = &s_tB[0][0];
  for (int i = 1; i < LN_; ++i) {
    #pragma unroll
    for (int j = 0; j < 8; ++j)
      #pragma unroll
      for (int r = 0; r < 8; ++r) acc[j][r] = 0.f;
    gemm_run<8>(src, H_, wmidb + (size_t)(i - 1) * HP_ * H_, H_,
                H_/32, acc, mblk, colgrp, lane);
    const float* biasp = sm_mid_b + (size_t)(i - 1) * H_;
    const float* bmi   = bm + (size_t)i * H_;
    #pragma unroll
    for (int j = 0; j < 8; ++j) {
      const int nb = colgrp + 4 * j;
      if (nb < H_/16) {
        const int colb = nb * 16 + rsel;
        const float bias = biasp[colb];
        const float g    = bmi[colb];
        #pragma unroll
        for (int r = 0; r < 8; ++r)
          dst[(size_t)(rbase + r) * H_ + colb] = f2bf_bits((acc[j][r] + bias) * g);
      }
    }
    __syncthreads();
    unsigned short* tmp = (unsigned short*)src; src = dst; dst = tmp;
  }
  // final temp now in `src` (== s_tA); `dst` (== s_tB) is free

  // ---------------- Last GEMM: sigmoid(temp @ sm_last_w.T + b) * xn --------
  #pragma unroll
  for (int j = 0; j < 8; ++j)
    #pragma unroll
    for (int r = 0; r < 8; ++r) acc[j][r] = 0.f;
  gemm_run<5>(src, H_, wlastb, H_, H_/32, acc, mblk, colgrp, lane);
  #pragma unroll
  for (int j = 0; j < 5; ++j) {            // 20 tiles == 4 colgrps * 5: all valid
    const int nb = colgrp + 4 * j;
    const int colb = nb * 16 + rsel;
    const float bias = sm_last_b[colb];
    #pragma unroll
    for (int r = 0; r < 8; ++r) {
      const float m  = 1.f / (1.f + __expf(-(acc[j][r] + bias)));
      const float xg = s_xn[rbase + r][colb] * m;
      dst[(size_t)(rbase + r) * H_ + colb] = f2bf_bits(xg);  // xg bf16, ld=H_
    }
  }
  __syncthreads();

  // ---------------- Head GEMM: xg @ head_w.T + head_b ----------------------
  v8f acch;
  #pragma unroll
  for (int r = 0; r < 8; ++r) acch[r] = 0.f;
  gemm_run<1>(dst, H_, wheadb, E_, E_/32, &acch, mblk, colgrp, lane);
  if (colgrp < C_/16) {                    // colgrps 2,3 hit zero-pad rows
    const int colb = colgrp * 16 + rsel;
    const float bias = head_b[colb];
    #pragma unroll
    for (int r = 0; r < 8; ++r)
      out[(size_t)(t0 + rbase + r) * C_ + colb] = acch[r] + bias;
  }
}

// ---------------------------------------------------------------------------
extern "C" void kernel_launch(void* const* d_in, const int* in_sizes, int n_in,
                              void* d_out, int out_size, void* d_ws, size_t ws_size,
                              hipStream_t stream) {
  (void)in_sizes; (void)n_in; (void)out_size; (void)ws_size;

  const float* x         = (const float*)d_in[0];
  const float* snr       = (const float*)d_in[1];
  const float* norm_w    = (const float*)d_in[2];
  const float* norm_b    = (const float*)d_in[3];
  const float* sm0_w     = (const float*)d_in[4];
  const float* sm0_b     = (const float*)d_in[5];
  const float* sm_mid_w  = (const float*)d_in[6];
  const float* sm_mid_b  = (const float*)d_in[7];
  const float* sm_last_w = (const float*)d_in[8];
  const float* sm_last_b = (const float*)d_in[9];
  const float* bm_w1     = (const float*)d_in[10];
  const float* bm_b1     = (const float*)d_in[11];
  const float* bm_w2     = (const float*)d_in[12];
  const float* bm_b2     = (const float*)d_in[13];
  const float* bm_w3     = (const float*)d_in[14];
  const float* bm_b3     = (const float*)d_in[15];
  const float* head_w    = (const float*)d_in[16];
  const float* head_b    = (const float*)d_in[17];
  float* out = (float*)d_out;

  // workspace layout (padded bf16 weights)
  char* ws = (char*)d_ws;
  float* bm = (float*)ws;                                     // 7*480 f32
  unsigned short* w0b    = (unsigned short*)(ws + 16384);     // HP_ x E
  unsigned short* wmidb  = w0b   + (size_t)HP_ * E_;          // 6 x HP_ x H
  unsigned short* wlastb = wmidb + (size_t)(LN_ - 1) * HP_ * H_; // E x H
  unsigned short* wheadb = wlastb + (size_t)E_ * H_;          // CP_ x E

  const int T = 256;
  const int n0 = HP_ * E_;
  const int n1 = (LN_ - 1) * HP_ * H_;
  const int n2 = E_ * H_;
  const int n3 = CP_ * E_;
  cvt_pad_kernel<<<(n0 + T - 1) / T, T, 0, stream>>>(sm0_w,     w0b,    H_, HP_, E_, 1);
  cvt_pad_kernel<<<(n1 + T - 1) / T, T, 0, stream>>>(sm_mid_w,  wmidb,  H_, HP_, H_, LN_ - 1);
  cvt_pad_kernel<<<(n2 + T - 1) / T, T, 0, stream>>>(sm_last_w, wlastb, E_, E_,  H_, 1);
  cvt_pad_kernel<<<(n3 + T - 1) / T, T, 0, stream>>>(head_w,    wheadb, C_, CP_, E_, 1);

  modulator_kernel<<<1, 512, 0, stream>>>(snr, bm_w1, bm_b1, bm_w2, bm_b2,
                                          bm_w3, bm_b3, bm);

  fused_encoder_kernel<<<NTOK / TM, 256, 0, stream>>>(
      x, norm_w, norm_b,
      w0b, sm0_b, wmidb, sm_mid_b, wlastb, sm_last_b, wheadb, head_b,
      bm, out);
}
